// ChitChat_30983894073583
// MI455X (gfx1250) — matrix operand order repaired and checked by reading, not compiled, not run on hardware
//
#include <hip/hip_runtime.h>
#include <hip/hip_bf16.h>
#include <stdint.h>

// Shapes
#define BB 64
#define SS 64
#define TT 64
#define VV 20000
#define EE 100
#define UU 300
#define FU 1200      // 4*U
#define UP 304       // U padded to 19 tiles of 16
#define KPD 320      // dense K padded to 10 chunks of 32

typedef __attribute__((ext_vector_type(2)))  float    v2f;
typedef __attribute__((ext_vector_type(8)))  float    v8f;
typedef __attribute__((ext_vector_type(16))) _Float16 v16h;

__device__ __forceinline__ float sigmoidf_(float x) { return 1.f / (1.f + __expf(-x)); }

// ---------------------------------------------------------------------------
// Init: zero h double-buffer, c buffer, barrier counters, and the padded f16
// dense-A buffer (decoder writes only u<300; pad columns must stay zero).
// ---------------------------------------------------------------------------
__global__ void k_init(float* hbuf, float* cbuf, unsigned* ctr, _Float16* sf) {
    const int idx = blockIdx.x * 256 + threadIdx.x;
    const int NH = 2 * BB * UP;   // 38912
    const int NC = BB * UP;       // 19456
    if (idx < NH) hbuf[idx] = 0.f;
    else if (idx < NH + NC) cbuf[idx - NH] = 0.f;
    if (idx < 2) ctr[idx] = 0u;
    if (idx < 4096 * KPD) sf[idx] = (_Float16)0.f;
}

// ---------------------------------------------------------------------------
// Input projection: XK[t][b][0:1200] = X[(t,b),0:100] @ W[100][1200] + bias
// mode 0: encoder (X row gathered from embedding via inputs[b*S+s])
// mode 1: decoder (X row = decoder_inputs[(b*T+t)*E])
// One wave per 16x16 output tile; f32 WMMA 16x16x4.
// ---------------------------------------------------------------------------
__global__ __launch_bounds__(256) void k_pregemm(
    const float* __restrict__ xrows, const int* __restrict__ gidx,
    const float* __restrict__ Wk, const float* __restrict__ bias,
    float* __restrict__ out, int mode)
{
    const int lane = threadIdx.x & 31;
    const int w    = threadIdx.x >> 5;
    const int mt   = blockIdx.x;             // 0..255
    const int nt   = blockIdx.y * 8 + w;     // 0..79 (75 valid)
    if (nt >= FU / 16) return;               // wave-uniform

    const int ml = lane & 15;                // m within tile / n within tile
    const int kh = lane >> 4;                // K-half selector
    const int R  = mt * 16 + ml;             // GEMM row = t*64 + b

    const float* arow;
    if (mode == 0) { int s = R >> 6, b = R & 63; arow = xrows + (size_t)gidx[b * SS + s] * EE; }
    else           { int t = R >> 6, b = R & 63; arow = xrows + ((size_t)(b * TT + t)) * EE; }

    const int ncol0 = nt * 16;
    v8f acc = {};
    for (int k0 = 0; k0 < EE; k0 += 4) {
        const int kk = k0 + 2 * kh;
        v2f af; af.x = arow[kk]; af.y = arow[kk + 1];
        const float* wp = Wk + (size_t)kk * FU + ncol0 + ml;
        v2f bf; bf.x = wp[0]; bf.y = wp[FU];
        acc = __builtin_amdgcn_wmma_f32_16x16x4_f32(false, af, false, bf,
                                                    (short)0, acc, false, false);
    }
    const float bv = bias[ncol0 + ml];
#pragma unroll
    for (int r = 0; r < 8; ++r) {
        const int Rr = mt * 16 + r + kh * 8;
        out[(size_t)Rr * FU + ncol0 + ml] = acc[r] + bv;
    }
}

// ---------------------------------------------------------------------------
// Persistent LSTM scan. 19 workgroups; block `blk` owns u in [16*blk,16*blk+16)
// across all 4 gates. Recurrent weights (transposed, padded) + cell state in
// LDS; h double-buffered in global, one atomic grid barrier per step.
// Decoder emits hidden states directly as f16 into the dense-A buffer.
// ---------------------------------------------------------------------------
__global__ __launch_bounds__(256) void k_lstm(
    const float* __restrict__ xk,    // [T][64][1200] precomputed x@W+bias
    const float* __restrict__ rec,   // [300][1200]
    float* __restrict__ hbuf,        // 2 x [64][304]
    float* __restrict__ cbuf,        // [64][304]
    _Float16* __restrict__ seq16,    // [T*64][320] f16 or nullptr
    unsigned* __restrict__ ctr, int nsteps, int write_c)
{
    extern __shared__ __align__(16) char smem[];
    float* w_lds = (float*)smem;                               // [64][304] transposed weights
    float* z_lds = (float*)(smem + 64 * UP * 4);               // [64][64] z slab
    float* c_lds = (float*)(smem + 64 * UP * 4 + 64 * 64 * 4); // [64][16] cell state

    const int tid  = threadIdx.x, lane = tid & 31, w = tid >> 5;
    const int blk  = blockIdx.x, u0 = blk * 16;

    // Load transposed, zero-padded recurrent weight slab: w_lds[g*16+j][k]
    for (int idx = tid; idx < 64 * UP; idx += 256) {
        const int ncol = idx / UP, k = idx % UP;
        const int g = ncol >> 4, j = ncol & 15, u = u0 + j;
        w_lds[idx] = (k < UU && u < UU) ? rec[(size_t)k * FU + g * UU + u] : 0.f;
    }
    for (int idx = tid; idx < 64 * 16; idx += 256) {
        const int m = idx >> 4, j = idx & 15;
        c_lds[idx] = cbuf[m * UP + u0 + j];
    }
    __syncthreads();

    const int mt = w & 3;             // M tile (rows 16mt..16mt+15)
    const int g0 = (w >> 2) * 2, g1 = g0 + 1;
    const int ml = lane & 15, kh = lane >> 4;

    for (int t = 0; t < nsteps; ++t) {
        const float* hread  = hbuf + (t & 1) * (BB * UP);
        float*       hwrite = hbuf + ((t + 1) & 1) * (BB * UP);
        const float* xkt    = xk + (size_t)t * BB * FU;

        // z_rec = h @ rec  (M=64, K=304, N=64 per block), f32 WMMA
        v8f acc0 = {}, acc1 = {};
        const float* hrow = hread + (mt * 16 + ml) * UP;
        const float* b0p  = w_lds + (g0 * 16 + ml) * UP;
        const float* b1p  = w_lds + (g1 * 16 + ml) * UP;
        for (int k0 = 0; k0 < UP; k0 += 4) {
            const int kk = k0 + 2 * kh;
            v2f af  = *(const v2f*)(hrow + kk);
            v2f bf0 = *(const v2f*)(b0p + kk);
            v2f bf1 = *(const v2f*)(b1p + kk);
            acc0 = __builtin_amdgcn_wmma_f32_16x16x4_f32(false, af, false, bf0,
                                                         (short)0, acc0, false, false);
            acc1 = __builtin_amdgcn_wmma_f32_16x16x4_f32(false, af, false, bf1,
                                                         (short)0, acc1, false, false);
        }
#pragma unroll
        for (int r = 0; r < 8; ++r) {
            const int m = mt * 16 + r + kh * 8;
            z_lds[m * 64 + g0 * 16 + ml] = acc0[r];
            z_lds[m * 64 + g1 * 16 + ml] = acc1[r];
        }
        __syncthreads();

        // Gate math for this block's 64x16 slab of u
#pragma unroll
        for (int q = 0; q < 4; ++q) {
            const int idx = tid + q * 256;              // 0..1023
            const int m = idx >> 4, j = idx & 15, u = u0 + j;
            if (u < UU) {
                const float* zr = z_lds + m * 64;
                const float* xr = xkt + m * FU;
                const float zi = zr[0 * 16 + j] + xr[0 * UU + u];
                const float zf = zr[1 * 16 + j] + xr[1 * UU + u];
                const float zg = zr[2 * 16 + j] + xr[2 * UU + u];
                const float zo = zr[3 * 16 + j] + xr[3 * UU + u];
                const float cn = sigmoidf_(zf) * c_lds[idx] + sigmoidf_(zi) * tanhf(zg);
                const float hn = sigmoidf_(zo) * tanhf(cn);
                c_lds[idx] = cn;
                hwrite[m * UP + u] = hn;
                if (seq16) seq16[((size_t)t * BB + m) * KPD + u] = (_Float16)hn;
                if (t + 1 < nsteps)
                    __builtin_prefetch(xr + BB * FU + u, 0, 0);   // next step's xk slab
            }
        }

        // Grid-wide barrier (h double-buffered, so one barrier per step)
        __threadfence();
        __syncthreads();
        if (tid == 0) {
            __hip_atomic_fetch_add(ctr, 1u, __ATOMIC_ACQ_REL, __HIP_MEMORY_SCOPE_AGENT);
            const unsigned target = (unsigned)(gridDim.x * (t + 1));
            while (__hip_atomic_load(ctr, __ATOMIC_ACQUIRE, __HIP_MEMORY_SCOPE_AGENT) < target)
                __builtin_amdgcn_s_sleep(1);
        }
        __syncthreads();
    }

    if (write_c) {
        for (int idx = tid; idx < 64 * 16; idx += 256) {
            const int m = idx >> 4, j = idx & 15, u = u0 + j;
            if (u < UU) cbuf[m * UP + u] = c_lds[idx];
        }
    }
}

// ---------------------------------------------------------------------------
// f32 -> f16 transpose + zero-pad of dense weights: dw[300][20000] -> wt[20000][320]
// ---------------------------------------------------------------------------
__global__ void k_cvt_wt(const float* __restrict__ dw, _Float16* __restrict__ wt) {
    size_t idx = (size_t)blockIdx.x * 256 + threadIdx.x;
    if (idx >= (size_t)VV * KPD) return;
    const int u = (int)(idx % KPD);
    const size_t v = idx / KPD;
    wt[idx] = (_Float16)((u < UU) ? dw[(size_t)u * VV + v] : 0.f);
}

// ---------------------------------------------------------------------------
// Dense pass 1: logits = seq @ dense_w + b (f16 WMMA, f32 acc), write raw
// logits and fused online softmax stats. One block per 64-row slab (grid 64);
// each wave computes 4 M-tiles x 1 N-tile so every B fragment is reused 4x.
// ---------------------------------------------------------------------------
__global__ __launch_bounds__(256) void k_dense(
    const _Float16* __restrict__ A16, const _Float16* __restrict__ Wt,
    const float* __restrict__ db, float* __restrict__ out,
    float* __restrict__ rowmax, float* __restrict__ rowsum)
{
    extern __shared__ __align__(16) char smem[];
    _Float16* a_lds = (_Float16*)smem;                  // [64][320] = 40 KB
    float2*   stats = (float2*)(smem + 64 * KPD * 2);   // [8 waves][64 rows]

    const int tid = threadIdx.x, lane = tid & 31, w = tid >> 5;
    const size_t row0 = (size_t)blockIdx.x * 64;

    {   // stage 64-row A slab into LDS (uint4 vector copies)
        const uint4* src = (const uint4*)(A16 + row0 * KPD);
        uint4* dst = (uint4*)a_lds;
        for (int i = tid; i < 64 * KPD / 8; i += 256) dst[i] = src[i];
    }
    __syncthreads();

    const int ml = lane & 15, half = lane >> 4;
    float lmax[4][8], lsum[4][8];
#pragma unroll
    for (int mt = 0; mt < 4; ++mt)
#pragma unroll
        for (int r = 0; r < 8; ++r) { lmax[mt][r] = -3.0e38f; lsum[mt][r] = 0.f; }

    for (int ch = 0; ch < 157; ++ch) {
        const int nt = ch * 8 + w;
        if (nt < VV / 16) {                       // wave-uniform
            const int ncol0 = nt * 16;
            v8f acc[4] = {{}, {}, {}, {}};
            for (int kc = 0; kc < 10; ++kc) {
                const int k0 = kc * 32;
                v16h bm;
                const _Float16* bp = Wt + (size_t)(ncol0 + ml) * KPD + k0 + 8 * half;
                ((uint4*)&bm)[0] = *(const uint4*)bp;
                ((uint4*)&bm)[1] = *(const uint4*)(bp + 16);
#pragma unroll
                for (int mt = 0; mt < 4; ++mt) {
                    v16h a;
                    const _Float16* ap = a_lds + (mt * 16 + ml) * KPD + k0 + 8 * half;
                    ((uint4*)&a)[0] = *(const uint4*)ap;
                    ((uint4*)&a)[1] = *(const uint4*)(ap + 16);
                    acc[mt] = __builtin_amdgcn_wmma_f32_16x16x32_f16(false, a, false, bm,
                                                                     (short)0, acc[mt],
                                                                     false, false);
                }
            }
            const float bv = db[ncol0 + ml];
#pragma unroll
            for (int mt = 0; mt < 4; ++mt)
#pragma unroll
                for (int r = 0; r < 8; ++r) {
                    const float x = acc[mt][r] + bv;
                    const int R = (int)row0 + mt * 16 + r + half * 8;   // R = t*64 + b
                    const int b = R & 63, t = R >> 6;
                    out[((size_t)(b * TT + t)) * VV + ncol0 + ml] = x;
                    const float nm = fmaxf(lmax[mt][r], x);
                    lsum[mt][r] = lsum[mt][r] * __expf(lmax[mt][r] - nm) + __expf(x - nm);
                    lmax[mt][r] = nm;
                }
        }
    }

    // reduce stats across the 16 lanes of each half-wave (N direction)
#pragma unroll
    for (int mt = 0; mt < 4; ++mt)
#pragma unroll
        for (int r = 0; r < 8; ++r) {
            for (int mask = 1; mask < 16; mask <<= 1) {
                const float om = __shfl_xor(lmax[mt][r], mask, 32);
                const float os = __shfl_xor(lsum[mt][r], mask, 32);
                const float nm = fmaxf(lmax[mt][r], om);
                lsum[mt][r] = lsum[mt][r] * __expf(lmax[mt][r] - nm) + os * __expf(om - nm);
                lmax[mt][r] = nm;
            }
        }
    if (ml == 0) {
#pragma unroll
        for (int mt = 0; mt < 4; ++mt)
#pragma unroll
            for (int r = 0; r < 8; ++r)
                stats[w * 64 + mt * 16 + r + half * 8] = make_float2(lmax[mt][r], lsum[mt][r]);
    }
    __syncthreads();
    if (tid < 64) {   // merge the 8 waves' partials per row
        float M = -3.0e38f, S = 0.f;
        for (int ww = 0; ww < 8; ++ww) {
            const float2 p = stats[ww * 64 + tid];
            const float nm = fmaxf(M, p.x);
            S = S * __expf(M - nm) + p.y * __expf(p.x - nm);
            M = nm;
        }
        rowmax[row0 + tid] = M;
        rowsum[row0 + tid] = S;
    }
}

// ---------------------------------------------------------------------------
// Dense pass 2: normalize logits -> softmax
// ---------------------------------------------------------------------------
__global__ void k_softmax(float* __restrict__ out, const float* __restrict__ rowmax,
                          const float* __restrict__ rowsum)
{
    const size_t idx = (size_t)blockIdx.x * 256 + threadIdx.x;
    if (idx >= (size_t)BB * TT * VV) return;
    const size_t row = idx / VV;               // = b*64 + t
    const int b = (int)(row >> 6), t = (int)(row & 63);
    const int R = t * 64 + b;                  // stats row
    out[idx] = __expf(out[idx] - rowmax[R]) / rowsum[R];
}

// ---------------------------------------------------------------------------
extern "C" void kernel_launch(void* const* d_in, const int* in_sizes, int n_in,
                              void* d_out, int out_size, void* d_ws, size_t ws_size,
                              hipStream_t stream) {
    const int*   inputs     = (const int*)  d_in[0];
    const float* dec_inputs = (const float*)d_in[1];
    const float* embedding  = (const float*)d_in[2];
    const float* enc_kernel = (const float*)d_in[3];
    const float* enc_rec    = (const float*)d_in[4];
    const float* enc_bias   = (const float*)d_in[5];
    const float* dec_kernel = (const float*)d_in[6];
    const float* dec_rec    = (const float*)d_in[7];
    const float* dec_bias   = (const float*)d_in[8];
    const float* dense_w    = (const float*)d_in[9];
    const float* dense_b    = (const float*)d_in[10];
    float* out = (float*)d_out;

    uint8_t* ws = (uint8_t*)d_ws;
    size_t off = 0;
    auto alloc = [&](size_t bytes) -> void* {
        void* p = ws + off;
        off += (bytes + 1023) & ~(size_t)1023;
        return p;
    };
    float*     xk_enc = (float*)    alloc((size_t)SS * BB * FU * 4);   // 19.7 MB
    float*     xk_dec = (float*)    alloc((size_t)TT * BB * FU * 4);   // 19.7 MB
    float*     hbuf   = (float*)    alloc((size_t)2 * BB * UP * 4);
    float*     cbuf   = (float*)    alloc((size_t)BB * UP * 4);
    _Float16*  seqf16 = (_Float16*) alloc((size_t)4096 * KPD * 2);     // 2.6 MB
    _Float16*  wt16   = (_Float16*) alloc((size_t)VV * KPD * 2);       // 12.8 MB
    float*     rowmax = (float*)    alloc(4096 * 4);
    float*     rowsum = (float*)    alloc(4096 * 4);
    unsigned*  ctr    = (unsigned*) alloc(256);

    // init h/c/counters + zero the padded dense-A f16 buffer
    k_init<<<(4096 * KPD + 255) / 256, 256, 0, stream>>>(hbuf, cbuf, ctr, seqf16);

    // input projections (parallel over all timesteps), f32 WMMA
    k_pregemm<<<dim3(256, 10), 256, 0, stream>>>(embedding, inputs, enc_kernel, enc_bias, xk_enc, 0);
    k_pregemm<<<dim3(256, 10), 256, 0, stream>>>(dec_inputs, nullptr, dec_kernel, dec_bias, xk_dec, 1);

    const size_t lstm_lds = (size_t)64 * UP * 4 + 64 * 64 * 4 + 64 * 16 * 4;  // 96 KB
    // encoder scan: keep final h (in hbuf parity 0) and c (written back to cbuf)
    k_lstm<<<UP / 16, 256, lstm_lds, stream>>>(xk_enc, enc_rec, hbuf, cbuf, nullptr, ctr, SS, 1);
    // decoder scan: starts from encoder state, emits f16 sequence for the dense GEMM
    k_lstm<<<UP / 16, 256, lstm_lds, stream>>>(xk_dec, dec_rec, hbuf, cbuf, seqf16, ctr + 1, TT, 0);

    // dense layer in f16 WMMA + fused online softmax stats
    k_cvt_wt<<<(int)(((size_t)VV * KPD + 255) / 256), 256, 0, stream>>>(dense_w, wt16);
    const size_t dense_lds = (size_t)64 * KPD * 2 + 8 * 64 * sizeof(float2);  // 44 KB
    k_dense<<<64, 256, dense_lds, stream>>>(seqf16, wt16, dense_b, out, rowmax, rowsum);
    k_softmax<<<(int)(((size_t)BB * TT * VV + 255) / 256), 256, 0, stream>>>(out, rowmax, rowsum);
}
